// DecoderBlock_17265768530695
// MI455X (gfx1250) — matrix operand order repaired and checked by reading, not compiled, not run on hardware
//
#include <hip/hip_runtime.h>
#include <hip/hip_bf16.h>

// ---------------------------------------------------------------------------
// DecoderBlock on MI455X (gfx1250): bf16 WMMA pipeline with async-to-LDS
// double-buffered staging (GLOBAL_LOAD_ASYNC_TO_LDS_B128 / ASYNCcnt).
// ---------------------------------------------------------------------------

typedef __attribute__((ext_vector_type(16))) __bf16 v16bf;
typedef __attribute__((ext_vector_type(8)))  float  v8f;

#define HEADS   4
#define SEQ     2048
#define DMODEL  1024
#define KSZ     1024
#define ZDIM    (HEADS * KSZ)      // 4096
#define BATCHN  2
#define MROWS   (BATCHN * SEQ)     // 4096

// ---------------------------------------------------------------------------
// Weight standardization (per-column over axis 0, population variance),
// written TRANSPOSED so every GEMM B operand is N-major (K contiguous).
//   Ws_t[n][k] = (W[k][n] - mu_n) * rsqrt(var_n + eps)
// ---------------------------------------------------------------------------
__global__ __launch_bounds__(256)
void ws_kernel(const float* __restrict__ W, __bf16* __restrict__ Wst, int R, int C)
{
    int col = blockIdx.x * blockDim.x + threadIdx.x;
    if (col >= C) return;
    float s = 0.f, s2 = 0.f;
    for (int r = 0; r < R; ++r) {
        float w = W[(size_t)r * C + col];
        s += w; s2 += w * w;
    }
    float mu   = s / (float)R;
    float var  = s2 / (float)R - mu * mu;
    float rstd = rsqrtf(var + 1e-5f);
    __bf16* out = Wst + (size_t)col * R;
    for (int r = 0; r < R; ++r) {
        float w = W[(size_t)r * C + col];
        out[r] = (__bf16)((w - mu) * rstd);
    }
}

// ---------------------------------------------------------------------------
// RoPE + row statistics: xr (bf16), sum(x) and sum(xr) per row.
// ---------------------------------------------------------------------------
__global__ __launch_bounds__(256)
void rope_stats_kernel(const float* __restrict__ x, __bf16* __restrict__ xr,
                       float* __restrict__ xsum, float* __restrict__ xrsum)
{
    __shared__ float s0[256], s1[256];
    const int row = blockIdx.x;           // 0..MROWS-1
    const int t   = row & (SEQ - 1);
    const float* xp = x + (size_t)row * DMODEL;
    __bf16* rp = xr + (size_t)row * DMODEL;

    const float kLogScale = 9.210340371976184f / 512.f;  // ln(10000)/half
    float sx = 0.f, sr = 0.f;
    for (int i = threadIdx.x; i < 512; i += 256) {
        float x1 = xp[i], x2 = xp[i + 512];
        float ang = (float)t * __expf(-(float)i * kLogScale);
        float sn = __sinf(ang), cs = __cosf(ang);
        float r1 = x1 * cs - x2 * sn;
        float r2 = x2 * cs + x1 * sn;
        rp[i]       = (__bf16)r1;
        rp[i + 512] = (__bf16)r2;
        sx += x1 + x2;
        sr += r1 + r2;
    }
    s0[threadIdx.x] = sx; s1[threadIdx.x] = sr;
    __syncthreads();
    for (int off = 128; off > 0; off >>= 1) {
        if (threadIdx.x < off) {
            s0[threadIdx.x] += s0[threadIdx.x + off];
            s1[threadIdx.x] += s1[threadIdx.x + off];
        }
        __syncthreads();
    }
    if (threadIdx.x == 0) { xsum[row] = s0[0]; xrsum[row] = s1[0]; }
}

// ---------------------------------------------------------------------------
// Causal-masked row softmax over f32 logits -> bf16 attention weights.
// ---------------------------------------------------------------------------
__global__ __launch_bounds__(256)
void softmax_kernel(const float* __restrict__ logits, __bf16* __restrict__ attn)
{
    __shared__ float sd[256];
    const int row   = blockIdx.x;         // over B*H*SEQ
    const int t     = row & (SEQ - 1);
    const int valid = t + 1;
    const float* L = logits + (size_t)row * SEQ;
    __bf16*      O = attn   + (size_t)row * SEQ;

    float v[8];
    float mx = -1e30f;
#pragma unroll
    for (int j = 0; j < 8; ++j) {
        int c = threadIdx.x + j * 256;
        v[j] = L[c];
        if (c < valid) mx = fmaxf(mx, v[j]);
    }
    sd[threadIdx.x] = mx; __syncthreads();
    for (int off = 128; off > 0; off >>= 1) {
        if (threadIdx.x < off) sd[threadIdx.x] = fmaxf(sd[threadIdx.x], sd[threadIdx.x + off]);
        __syncthreads();
    }
    mx = sd[0];
    __syncthreads();

    float sum = 0.f;
#pragma unroll
    for (int j = 0; j < 8; ++j) {
        int c = threadIdx.x + j * 256;
        float e = (c < valid) ? __expf(v[j] - mx) : 0.f;
        v[j] = e; sum += e;
    }
    sd[threadIdx.x] = sum; __syncthreads();
    for (int off = 128; off > 0; off >>= 1) {
        if (threadIdx.x < off) sd[threadIdx.x] += sd[threadIdx.x + off];
        __syncthreads();
    }
    float inv = 1.f / sd[0];
#pragma unroll
    for (int j = 0; j < 8; ++j)
        O[threadIdx.x + j * 256] = (__bf16)(v[j] * inv);
}

// ---------------------------------------------------------------------------
// outer = sum(x) * sum(xr) * attended / HEADS  -> bf16 for the final GEMM.
// ---------------------------------------------------------------------------
__global__ __launch_bounds__(256)
void outer_kernel(const float* __restrict__ att, const float* __restrict__ xs,
                  const float* __restrict__ xrs, __bf16* __restrict__ outp)
{
    const size_t total = (size_t)MROWS * ZDIM;
    for (size_t i = (size_t)blockIdx.x * blockDim.x + threadIdx.x; i < total;
         i += (size_t)gridDim.x * blockDim.x) {
        size_t row = i >> 12;  // / ZDIM
        outp[i] = (__bf16)(att[i] * (xs[row] * xrs[row] * 0.25f));
    }
}

// ---------------------------------------------------------------------------
// vh (4096 x 4096, [(b,t)][(h,dim)]) -> vh_t ([(b,h,dim)][t], rows of SEQ)
// LDS-tiled 64x64 transpose so the PV GEMM's B operand is N-major.
// ---------------------------------------------------------------------------
__global__ __launch_bounds__(256)
void transpose_vh_kernel(const __bf16* __restrict__ vh, __bf16* __restrict__ vht)
{
    __shared__ __bf16 tile[64][65];
    const int col0 = blockIdx.x * 64;   // over ZDIM (h*KSZ+dim)
    const int row0 = blockIdx.y * 64;   // over MROWS (b*SEQ+t)
    for (int i = threadIdx.x; i < 64 * 64; i += 256) {
        int r = i >> 6, c = i & 63;
        tile[r][c] = vh[(size_t)(row0 + r) * ZDIM + col0 + c];
    }
    __syncthreads();
    for (int i = threadIdx.x; i < 64 * 64; i += 256) {
        int r = i >> 6, c = i & 63;
        int gcol = col0 + r;            // h*KSZ + dim
        int grow = row0 + c;            // b*SEQ + t
        int b = grow >> 11, t = grow & (SEQ - 1);
        vht[((size_t)b * ZDIM + gcol) * SEQ + t] = tile[r][c];
    }
}

// ---------------------------------------------------------------------------
// Tiled bf16 WMMA GEMM:  C = alpha * A(MxK) @ B + bias
//   B operand is always N-major in memory (rows = n, K contiguous).
//   Block tile 128x128x32, 256 threads = 8 wave32 waves (4 M x 2 N),
//   each wave: 32x64 = 2x4 tiles of v_wmma_f32_16x16x32_bf16.
//   Double-buffered LDS staged with GLOBAL_LOAD_ASYNC_TO_LDS_B128;
//   completion via s_wait_asynccnt + workgroup barrier (ISA ch.10 / ASYNCcnt).
//   Fragment loads are contiguous 16B ds_load_b128 matching ISA 7.12.2:
//     A lane l: row = l&15, K = (l>>4)*8 + {0..7} and 16 + (l>>4)*8 + {0..7}
//     B lane l: col = l&15, K = (l>>4)*8 + {0..7} and 16 + (l>>4)*8 + {0..7}
//     C lane l: col = l&15, vgpr r -> row = r + (l>>4)*8
// ---------------------------------------------------------------------------
template<bool OUT_BF16>
__global__ __launch_bounds__(256)
void gemm_wmma(const __bf16* __restrict__ A, const __bf16* __restrict__ Bt,
               const float* __restrict__ bias, void* __restrict__ Cout,
               int M, int N, int K, int lda, int ldb, int ldc, float alpha,
               int batchH,
               long long sAb, long long sAh,
               long long sBb, long long sBh,
               long long sCb, long long sCh)
{
    constexpr int BM = 128, BN = 128, BK = 32, LDSW = 40;  // LDS row stride (bf16)
    __shared__ __align__(16) __bf16 As[2][BM * LDSW];
    __shared__ __align__(16) __bf16 Bs[2][BN * LDSW];

    const int tid  = threadIdx.x;
    const int lane = tid & 31;
    const int wave = tid >> 5;
    const int waveM = wave & 3;      // 4 strips of 32 rows
    const int waveN = wave >> 2;     // 2 strips of 64 cols
    const int l16 = lane & 15;
    const int lh  = lane >> 4;

    const int m0 = blockIdx.y * BM;
    const int n0 = blockIdx.x * BN;
    const int bz = blockIdx.z;
    const int bb = bz / batchH;
    const int hh = bz % batchH;
    A  += (long long)bb * sAb + (long long)hh * sAh;
    Bt += (long long)bb * sBb + (long long)hh * sBh;
    const long long cOff = (long long)bb * sCb + (long long)hh * sCh;

    // Async stage one 128x32 A tile + 128x32 B tile into LDS buffer `buf`.
    // 512 16B chunks per tile, 2 per thread per tile, all via async-to-LDS.
    auto stage = [&](int k0, int buf) {
#pragma unroll
        for (int ii = 0; ii < 2; ++ii) {
            const int c  = tid + ii * 256;
            const int r  = c >> 2;
            const int kc = (c & 3) * 8;
            const unsigned long long ga =
                (unsigned long long)(A + (long long)(m0 + r) * lda + k0 + kc);
            const unsigned la = (unsigned)(unsigned long long)&As[buf][r * LDSW + kc];
            asm volatile("global_load_async_to_lds_b128 %0, %1, off"
                         :: "v"(la), "v"(ga) : "memory");
            const unsigned long long gb =
                (unsigned long long)(Bt + (long long)(n0 + r) * ldb + k0 + kc);
            const unsigned lb = (unsigned)(unsigned long long)&Bs[buf][r * LDSW + kc];
            asm volatile("global_load_async_to_lds_b128 %0, %1, off"
                         :: "v"(lb), "v"(gb) : "memory");
        }
    };

    v8f acc[2][4] = {};

    stage(0, 0);
    asm volatile("s_wait_asynccnt 0x0" ::: "memory");
    __syncthreads();

    int cur = 0;
    for (int k0 = 0; k0 < K; k0 += BK) {
        if (k0 + BK < K) stage(k0 + BK, cur ^ 1);   // prefetch next tile async

        // ---- fragments + 8 WMMAs on current buffer
        v16bf afr[2], bfr[4];
#pragma unroll
        for (int mi = 0; mi < 2; ++mi) {
            const int r = waveM * 32 + mi * 16 + l16;
            uint4* dst = (uint4*)&afr[mi];
            dst[0] = *(const uint4*)(&As[cur][r * LDSW + lh * 8]);
            dst[1] = *(const uint4*)(&As[cur][r * LDSW + 16 + lh * 8]);
        }
#pragma unroll
        for (int ni = 0; ni < 4; ++ni) {
            const int nn = waveN * 64 + ni * 16 + l16;
            uint4* dst = (uint4*)&bfr[ni];
            dst[0] = *(const uint4*)(&Bs[cur][nn * LDSW + lh * 8]);
            dst[1] = *(const uint4*)(&Bs[cur][nn * LDSW + 16 + lh * 8]);
        }
#pragma unroll
        for (int mi = 0; mi < 2; ++mi)
#pragma unroll
            for (int ni = 0; ni < 4; ++ni)
                acc[mi][ni] = __builtin_amdgcn_wmma_f32_16x16x32_bf16(
                    false, afr[mi], false, bfr[ni], (short)0, acc[mi][ni],
                    false, false);

        // all our async writes (next tile) done + all waves done reading cur
        asm volatile("s_wait_asynccnt 0x0" ::: "memory");
        __syncthreads();
        cur ^= 1;
    }

    // ---- epilogue: alpha scale + bias, f32 or bf16 store
    float*  Cf = (float*)Cout + cOff;
    __bf16* Cb = (__bf16*)Cout + cOff;
#pragma unroll
    for (int mi = 0; mi < 2; ++mi) {
#pragma unroll
        for (int ni = 0; ni < 4; ++ni) {
            const int nn = n0 + waveN * 64 + ni * 16 + l16;
            const int mbase = m0 + waveM * 32 + mi * 16 + lh * 8;
            const float bv = bias ? bias[nn] : 0.f;
#pragma unroll
            for (int r = 0; r < 8; ++r) {
                float val = acc[mi][ni][r] * alpha + bv;
                long long off = (long long)(mbase + r) * ldc + nn;
                if (OUT_BF16) Cb[off] = (__bf16)val;
                else          Cf[off] = val;
            }
        }
    }
}

// ---------------------------------------------------------------------------
// Host orchestration
// ---------------------------------------------------------------------------
extern "C" void kernel_launch(void* const* d_in, const int* in_sizes, int n_in,
                              void* d_out, int out_size, void* d_ws, size_t ws_size,
                              hipStream_t stream)
{
    const float* x   = (const float*)d_in[0];
    const float* wq  = (const float*)d_in[1];
    const float* bq  = (const float*)d_in[2];
    const float* wk  = (const float*)d_in[3];
    const float* bk  = (const float*)d_in[4];
    const float* wv  = (const float*)d_in[5];
    const float* bv  = (const float*)d_in[6];
    const float* mwq = (const float*)d_in[7];
    const float* mbq = (const float*)d_in[8];
    const float* mwk = (const float*)d_in[9];
    const float* mbk = (const float*)d_in[10];
    const float* mwv = (const float*)d_in[11];
    const float* mbv = (const float*)d_in[12];
    const float* mwo = (const float*)d_in[13];
    const float* mbo = (const float*)d_in[14];
    const float* wf  = (const float*)d_in[15];
    const float* bf_ = (const float*)d_in[16];
    float* out = (float*)d_out;

    // ---- carve workspace
    char* p = (char*)d_ws;
    auto alloc = [&](size_t bytes) {
        void* r = (void*)p;
        p += (bytes + 255) & ~(size_t)255;
        return r;
    };
    __bf16* xr    = (__bf16*)alloc((size_t)MROWS * DMODEL * 2);
    float*  xsum  = (float*) alloc((size_t)MROWS * 4);
    float*  xrsum = (float*) alloc((size_t)MROWS * 4);
    __bf16* wq_s  = (__bf16*)alloc((size_t)DMODEL * DMODEL * 2);   // [n][k]
    __bf16* wk_s  = (__bf16*)alloc((size_t)DMODEL * DMODEL * 2);
    __bf16* wv_s  = (__bf16*)alloc((size_t)DMODEL * DMODEL * 2);
    __bf16* mwq_s = (__bf16*)alloc((size_t)DMODEL * ZDIM * 2);     // [n=4096][k=1024]
    __bf16* mwk_s = (__bf16*)alloc((size_t)DMODEL * ZDIM * 2);
    __bf16* mwv_s = (__bf16*)alloc((size_t)DMODEL * ZDIM * 2);
    __bf16* mwo_s = (__bf16*)alloc((size_t)ZDIM * ZDIM * 2);       // [n=4096][k=4096]
    __bf16* wf_s  = (__bf16*)alloc((size_t)ZDIM * DMODEL * 2);     // [n=1024][k=4096]
    __bf16* qb    = (__bf16*)alloc((size_t)MROWS * DMODEL * 2);
    __bf16* kb    = (__bf16*)alloc((size_t)MROWS * DMODEL * 2);
    __bf16* vb    = (__bf16*)alloc((size_t)MROWS * DMODEL * 2);
    __bf16* qh    = (__bf16*)alloc((size_t)MROWS * ZDIM * 2);
    __bf16* kh    = (__bf16*)alloc((size_t)MROWS * ZDIM * 2);
    __bf16* vh    = (__bf16*)alloc((size_t)MROWS * ZDIM * 2);
    __bf16* vht   = (__bf16*)alloc((size_t)MROWS * ZDIM * 2);      // [(b,h,dim)][t]
    float*  logit = (float*) alloc((size_t)BATCHN * HEADS * SEQ * SEQ * 4);
    __bf16* attn  = (__bf16*)alloc((size_t)BATCHN * HEADS * SEQ * SEQ * 2);
    __bf16* ctx   = (__bf16*)alloc((size_t)MROWS * ZDIM * 2);
    float*  attf  = (float*) alloc((size_t)MROWS * ZDIM * 4);
    __bf16* outer = (__bf16*)alloc((size_t)MROWS * ZDIM * 2);

    // ---- weight standardization (fp32 -> standardized, transposed bf16)
    ws_kernel<<<DMODEL / 256, 256, 0, stream>>>(wq,  wq_s,  DMODEL, DMODEL);
    ws_kernel<<<DMODEL / 256, 256, 0, stream>>>(wk,  wk_s,  DMODEL, DMODEL);
    ws_kernel<<<DMODEL / 256, 256, 0, stream>>>(wv,  wv_s,  DMODEL, DMODEL);
    ws_kernel<<<ZDIM  / 256, 256, 0, stream>>>(mwq, mwq_s, DMODEL, ZDIM);
    ws_kernel<<<ZDIM  / 256, 256, 0, stream>>>(mwk, mwk_s, DMODEL, ZDIM);
    ws_kernel<<<ZDIM  / 256, 256, 0, stream>>>(mwv, mwv_s, DMODEL, ZDIM);
    ws_kernel<<<ZDIM  / 256, 256, 0, stream>>>(mwo, mwo_s, ZDIM,   ZDIM);
    ws_kernel<<<DMODEL / 256, 256, 0, stream>>>(wf,  wf_s,  ZDIM,   DMODEL);

    // ---- RoPE + row stats
    rope_stats_kernel<<<MROWS, 256, 0, stream>>>(x, xr, xsum, xrsum);

    // ---- q/k/v = xr @ ws(w{q,k,v}) + b       (M=4096, N=1024, K=1024)
    dim3 g1(DMODEL / 128, MROWS / 128, 1);
    gemm_wmma<true><<<g1, 256, 0, stream>>>(xr, wq_s, bq, qb,
        MROWS, DMODEL, DMODEL, DMODEL, DMODEL, DMODEL, 1.f, 1, 0, 0, 0, 0, 0, 0);
    gemm_wmma<true><<<g1, 256, 0, stream>>>(xr, wk_s, bk, kb,
        MROWS, DMODEL, DMODEL, DMODEL, DMODEL, DMODEL, 1.f, 1, 0, 0, 0, 0, 0, 0);
    gemm_wmma<true><<<g1, 256, 0, stream>>>(xr, wv_s, bv, vb,
        MROWS, DMODEL, DMODEL, DMODEL, DMODEL, DMODEL, 1.f, 1, 0, 0, 0, 0, 0, 0);

    // ---- qh/kh/vh = q/k/v @ ws(mw*) + mb*    (M=4096, N=4096, K=1024)
    dim3 g2(ZDIM / 128, MROWS / 128, 1);
    gemm_wmma<true><<<g2, 256, 0, stream>>>(qb, mwq_s, mbq, qh,
        MROWS, ZDIM, DMODEL, DMODEL, DMODEL, ZDIM, 1.f, 1, 0, 0, 0, 0, 0, 0);
    gemm_wmma<true><<<g2, 256, 0, stream>>>(kb, mwk_s, mbk, kh,
        MROWS, ZDIM, DMODEL, DMODEL, DMODEL, ZDIM, 1.f, 1, 0, 0, 0, 0, 0, 0);
    gemm_wmma<true><<<g2, 256, 0, stream>>>(vb, mwv_s, mbv, vh,
        MROWS, ZDIM, DMODEL, DMODEL, DMODEL, ZDIM, 1.f, 1, 0, 0, 0, 0, 0, 0);

    // ---- vh -> vh_t  ([(b,h,dim)][t]) for an N-major PV GEMM B operand
    dim3 gt(ZDIM / 64, MROWS / 64, 1);
    transpose_vh_kernel<<<gt, 256, 0, stream>>>(vh, vht);

    // ---- logits[b,h] = (qh @ kh^T) / 32      (per (b,h): 2048x2048, K=1024)
    dim3 gl(SEQ / 128, SEQ / 128, BATCHN * HEADS);
    const long long nz  = (long long)SEQ * ZDIM;
    const long long nn2 = (long long)SEQ * SEQ;
    gemm_wmma<false><<<gl, 256, 0, stream>>>(qh, kh, nullptr, logit,
        SEQ, SEQ, KSZ, ZDIM, ZDIM, SEQ, 0.03125f, HEADS,
        nz, KSZ, nz, KSZ, HEADS * nn2, nn2);

    // ---- causal softmax -> bf16 attn
    softmax_kernel<<<BATCHN * HEADS * SEQ, 256, 0, stream>>>(logit, attn);

    // ---- ctx[b,:,h,:] = attn @ vh            (per (b,h): 2048x1024, K=2048)
    dim3 gc(KSZ / 128, SEQ / 128, BATCHN * HEADS);
    gemm_wmma<true><<<gc, 256, 0, stream>>>(attn, vht, nullptr, ctx,
        SEQ, KSZ, SEQ, SEQ, SEQ, ZDIM, 1.f, HEADS,
        HEADS * nn2, nn2,
        (long long)ZDIM * SEQ, (long long)KSZ * SEQ,
        nz, KSZ);

    // ---- attended = ctx @ ws(mwo) + mbo      (M=4096, N=4096, K=4096) -> f32
    dim3 go(ZDIM / 128, MROWS / 128, 1);
    gemm_wmma<false><<<go, 256, 0, stream>>>(ctx, mwo_s, mbo, attf,
        MROWS, ZDIM, ZDIM, ZDIM, ZDIM, ZDIM, 1.f, 1, 0, 0, 0, 0, 0, 0);

    // ---- outer = sum(x) * sum(xr) * attended / HEADS  -> bf16
    outer_kernel<<<8192, 256, 0, stream>>>(attf, xsum, xrsum, outer);

    // ---- out = outer @ ws(wf) + bf           (M=4096, N=1024, K=4096) -> f32
    dim3 gf(DMODEL / 128, MROWS / 128, 1);
    gemm_wmma<false><<<gf, 256, 0, stream>>>(outer, wf_s, bf_, out,
        MROWS, DMODEL, ZDIM, ZDIM, ZDIM, DMODEL, 1.f, 1, 0, 0, 0, 0, 0, 0);
}
